// WindowAttention3D_7447473291275
// MI455X (gfx1250) — matrix operand order repaired and checked by reading, not compiled, not run on hardware
//
#include <hip/hip_runtime.h>

#define NTOK 98
#define NPAD 112
#define DIM  128
#define HEADS 4
#define HD   32
#define NWIN 64
#define SCALE 0.17677669529663687f   // 1/sqrt(32)

typedef __attribute__((ext_vector_type(16))) _Float16 v16h;
typedef __attribute__((ext_vector_type(4)))  _Float16 v4h;
typedef __attribute__((ext_vector_type(8)))  float    v8f;

// A-fragment (16-bit, 16x32): lane M = l16; slots 0..15 -> K = koff+16*hi+i
// (contiguous, matching B-fragments)
__device__ __forceinline__ v16h a_frag_lds(const _Float16* base, int row, int stride,
                                           int koff, int hi) {
  v16h a;
  const _Float16* p = base + row * stride + koff + 16 * hi;
#pragma unroll
  for (int i = 0; i < 8; ++i) a[i] = p[i];
#pragma unroll
  for (int i = 0; i < 8; ++i) a[8 + i] = p[8 + i];
  return a;
}

// B-fragment (16-bit, 32x16) from row-major f32 weight matrix W[rows][128]:
// B[k][n] = W[row][kbase + k_local], lane n = l16, k_local = 16*hi + i (contiguous)
__device__ __forceinline__ v16h b_frag_w(const float* W, int row, int kbase) {
  const float4* p4 = reinterpret_cast<const float4*>(W + (size_t)row * DIM + kbase);
  v16h b;
#pragma unroll
  for (int q = 0; q < 4; ++q) {
    float4 f = p4[q];
    b[4 * q + 0] = (_Float16)f.x; b[4 * q + 1] = (_Float16)f.y;
    b[4 * q + 2] = (_Float16)f.z; b[4 * q + 3] = (_Float16)f.w;
  }
  return b;
}

// precompute comb[iw][h][m][n] = bias_table[rel_idx[m][n]][h] + mask[iw][m][n]
__global__ void comb_kernel(const float* __restrict__ maskg,
                            const float* __restrict__ bias_t,
                            const int*   __restrict__ rel_idx,
                            float* __restrict__ comb) {
  int idx = blockIdx.x * 256 + threadIdx.x;
  if (idx >= NWIN * HEADS * NTOK * NTOK) return;
  int n = idx % NTOK;
  int t = idx / NTOK;
  int m = t % NTOK;  t /= NTOK;
  int h = t % HEADS;
  int iw = t / HEADS;
  comb[idx] = bias_t[rel_idx[m * NTOK + n] * HEADS + h] +
              maskg[(size_t)iw * NTOK * NTOK + m * NTOK + n];
}

template <bool USE_COMB>
__global__ __launch_bounds__(224)
void win_attn3d_kernel(const float* __restrict__ xg,
                       const float* __restrict__ maskg,
                       const float* __restrict__ qkv_w,
                       const float* __restrict__ qkv_b,
                       const float* __restrict__ proj_w,
                       const float* __restrict__ proj_b,
                       const float* __restrict__ bias_t,
                       const int*   __restrict__ rel_idx,
                       const float* __restrict__ comb,
                       float* __restrict__ outg) {
  __shared__ _Float16 xf[NPAD * DIM];     // 28672 B  x window, f16, row-major
  __shared__ _Float16 qf[NPAD * HD];      //  7168 B  Q_h (pre-scaled), row-major
  __shared__ _Float16 kf[NPAD * HD];      //  7168 B  K_h, row-major (B-frag contiguous)
  __shared__ _Float16 vt[HD * 128];       //  8192 B  V_h transposed [d][krow]
  __shared__ _Float16 scr[7][16 * HD];    //  7168 B  per-wave C->A bounce
  // total 58368 B static LDS

  const int w   = blockIdx.x;
  const int tid = threadIdx.x;
  const int wv  = tid >> 5;      // wave id == row tile 0..6
  const int ln  = tid & 31;
  const int l16 = ln & 15;
  const int hi  = ln >> 4;

  const float* xw = xg + (size_t)w * NTOK * DIM;

  // ---- stage x -> LDS f16 (float4 loads, packed 4-half stores); zero pad rows
  for (int idx = tid; idx < (NPAD * DIM) / 4; idx += 224) {
    int r = idx >> 5, c4 = idx & 31;
    float4 f = make_float4(0.f, 0.f, 0.f, 0.f);
    if (r < NTOK) f = *reinterpret_cast<const float4*>(xw + r * DIM + c4 * 4);
    v4h hv = { (_Float16)f.x, (_Float16)f.y, (_Float16)f.z, (_Float16)f.w };
    *reinterpret_cast<v4h*>(xf + idx * 4) = hv;
  }
  // zero V^T pad columns (k-rows 112..127), never written afterwards
  for (int idx = tid; idx < HD * 16; idx += 224)
    vt[(idx >> 4) * 128 + NPAD + (idx & 15)] = (_Float16)0.f;
  __syncthreads();

  v8f pacc[8];                       // 16x128 f32 projection accumulator (per wave)
#pragma unroll
  for (int c = 0; c < 8; ++c) pacc[c] = {};

  for (int h = 0; h < HEADS; ++h) {
    // ======== Phase A: Q_h/K_h/V_h = x @ qkv_w^T + b (WMMA, W streamed from L2)
    v16h ax[4];
#pragma unroll
    for (int kk = 0; kk < 4; ++kk)
      ax[kk] = a_frag_lds(xf, 16 * wv + l16, DIM, 32 * kk, hi);

#pragma unroll
    for (int mat = 0; mat < 3; ++mat) {
#pragma unroll
      for (int c2 = 0; c2 < 2; ++c2) {
        int col0 = mat * DIM + h * HD + c2 * 16;
        v8f acc = {};
#pragma unroll
        for (int kk = 0; kk < 4; ++kk) {
          v16h bw = b_frag_w(qkv_w, col0 + l16, 32 * kk + 16 * hi);
          acc = __builtin_amdgcn_wmma_f32_16x16x32_f16(false, ax[kk], false, bw,
                                                       (short)0, acc, false, false);
        }
        float bias = qkv_b[col0 + l16];
        int cih = c2 * 16 + l16;
#pragma unroll
        for (int j = 0; j < 8; ++j) {
          int m = 16 * wv + j + 8 * hi;
          float v = acc[j] + bias;
          if (mat == 0)      qf[m * HD + cih] = (_Float16)(v * SCALE);
          else if (mat == 1) kf[m * HD + cih] = (_Float16)v;
          else               vt[cih * 128 + m] = (_Float16)v;   // transposed
        }
      }
    }
    __syncthreads();

    // ======== Phase B: scores = Q_h @ K_h^T  (one WMMA per 16x16 tile, K=32=hd)
    v16h aq = a_frag_lds(qf, 16 * wv + l16, HD, 0, hi);
    v8f sc[7];
#pragma unroll
    for (int ct = 0; ct < 7; ++ct) {
      v16h bk;
      const _Float16* kp = kf + (16 * ct + l16) * HD + 16 * hi;
#pragma unroll
      for (int i = 0; i < 16; ++i) bk[i] = kp[i];
      v8f z = {};
      sc[ct] = __builtin_amdgcn_wmma_f32_16x16x32_f16(false, aq, false, bk,
                                                      (short)0, z, false, false);
    }
    // + (relative-position bias + window mask); pad -> -inf
    if constexpr (USE_COMB) {
      const float* cp = comb + ((size_t)(w % NWIN) * HEADS + h) * NTOK * NTOK;
#pragma unroll
      for (int ct = 0; ct < 7; ++ct) {
#pragma unroll
        for (int j = 0; j < 8; ++j) {
          int m = 16 * wv + j + 8 * hi;
          int n = 16 * ct + l16;
          if (m < NTOK && n < NTOK) sc[ct][j] = sc[ct][j] + cp[m * NTOK + n];
          else                      sc[ct][j] = -1e30f;
        }
      }
    } else {
      const float* mk = maskg + (size_t)(w % NWIN) * NTOK * NTOK;
#pragma unroll
      for (int ct = 0; ct < 7; ++ct) {
#pragma unroll
        for (int j = 0; j < 8; ++j) {
          int m = 16 * wv + j + 8 * hi;
          int n = 16 * ct + l16;
          if (m < NTOK && n < NTOK) {
            int id = rel_idx[m * NTOK + n];
            sc[ct][j] = sc[ct][j] + bias_t[id * HEADS + h] + mk[m * NTOK + n];
          } else {
            sc[ct][j] = -1e30f;
          }
        }
      }
    }
    // row softmax: rows live across 16 lanes of a half-wave in C-layout
    float rmax[8], rinv[8];
#pragma unroll
    for (int j = 0; j < 8; ++j) {
      float mx = sc[0][j];
#pragma unroll
      for (int ct = 1; ct < 7; ++ct) mx = fmaxf(mx, sc[ct][j]);
#pragma unroll
      for (int off = 8; off >= 1; off >>= 1)
        mx = fmaxf(mx, __shfl_xor(mx, off, 16));
      rmax[j] = mx;
    }
#pragma unroll
    for (int j = 0; j < 8; ++j) {
      float s = 0.f;
#pragma unroll
      for (int ct = 0; ct < 7; ++ct) {
        float e = __expf(sc[ct][j] - rmax[j]);
        sc[ct][j] = e;
        s += e;
      }
#pragma unroll
      for (int off = 8; off >= 1; off >>= 1)
        s += __shfl_xor(s, off, 16);
      rinv[j] = 1.f / s;
    }
#pragma unroll
    for (int ct = 0; ct < 7; ++ct)
#pragma unroll
      for (int j = 0; j < 8; ++j) sc[ct][j] = sc[ct][j] * rinv[j];

    // ======== Phase C: O_h = P @ V_h  (C->A via per-wave LDS bounce, wave-local)
    v8f oacc[2];
    oacc[0] = {}; oacc[1] = {};
    _Float16* sw = scr[wv];
#pragma unroll
    for (int kk = 0; kk < 4; ++kk) {
#pragma unroll
      for (int s = 0; s < 2; ++s) {
        int ct = 2 * kk + s;
#pragma unroll
        for (int j = 0; j < 8; ++j) {
          float p = (ct < 7) ? sc[ct][j] : 0.f;     // score cols 112..127 -> 0
          sw[(j + 8 * hi) * HD + s * 16 + l16] = (_Float16)p;
        }
      }
      v16h pa = a_frag_lds(sw, l16, HD, 0, hi);
#pragma unroll
      for (int c2 = 0; c2 < 2; ++c2) {
        v16h bv;
        const _Float16* vp = vt + (c2 * 16 + l16) * 128 + 32 * kk + 16 * hi;
#pragma unroll
        for (int i = 0; i < 16; ++i) bv[i] = vp[i];   // contiguous -> 2x ds_load_b128
        oacc[c2] = __builtin_amdgcn_wmma_f32_16x16x32_f16(false, pa, false, bv,
                                                          (short)0, oacc[c2], false, false);
      }
    }

    // ======== per-head projection: pacc += O_h @ proj_w[:, h*32:(h+1)*32]^T
#pragma unroll
    for (int c2 = 0; c2 < 2; ++c2)
#pragma unroll
      for (int j = 0; j < 8; ++j)
        sw[(j + 8 * hi) * HD + c2 * 16 + l16] = (_Float16)oacc[c2][j];
    v16h po = a_frag_lds(sw, l16, HD, 0, hi);
#pragma unroll
    for (int c = 0; c < 8; ++c) {
      v16h bp = b_frag_w(proj_w, c * 16 + l16, h * HD + 16 * hi);
      pacc[c] = __builtin_amdgcn_wmma_f32_16x16x32_f16(false, po, false, bp,
                                                       (short)0, pacc[c], false, false);
    }
    __syncthreads();   // q/k/vt reused by next head
  }

  // ---- epilogue: + proj_b, scatter valid rows (coalesced across lanes)
  float* ow = outg + (size_t)w * NTOK * DIM;
#pragma unroll
  for (int c = 0; c < 8; ++c) {
    int col = 16 * c + l16;
    float pb = proj_b[col];
#pragma unroll
    for (int j = 0; j < 8; ++j) {
      int m = 16 * wv + j + 8 * hi;
      if (m < NTOK) ow[m * DIM + col] = pacc[c][j] + pb;
    }
  }
}

extern "C" void kernel_launch(void* const* d_in, const int* in_sizes, int n_in,
                              void* d_out, int out_size, void* d_ws, size_t ws_size,
                              hipStream_t stream) {
  const float* x      = (const float*)d_in[0];
  const float* mask   = (const float*)d_in[1];
  const float* qkv_w  = (const float*)d_in[2];
  const float* qkv_b  = (const float*)d_in[3];
  const float* proj_w = (const float*)d_in[4];
  const float* proj_b = (const float*)d_in[5];
  const float* bias_t = (const float*)d_in[6];
  const int*   relidx = (const int*)d_in[7];
  float* out = (float*)d_out;
  (void)n_in; (void)out_size;

  int nwin = in_sizes[0] / (NTOK * DIM);   // 2048
  dim3 grid(nwin), block(224);

  // fused (bias+mask) table in workspace if it fits: 64*4*98*98 floats = 9.8 MB
  const size_t comb_elems = (size_t)NWIN * HEADS * NTOK * NTOK;
  if (d_ws != nullptr && ws_size >= comb_elems * sizeof(float)) {
    float* comb = (float*)d_ws;
    int blocks = (int)((comb_elems + 255) / 256);
    hipLaunchKernelGGL(comb_kernel, dim3(blocks), dim3(256), 0, stream,
                       mask, bias_t, relidx, comb);
    hipLaunchKernelGGL((win_attn3d_kernel<true>), grid, block, 0, stream,
                       x, mask, qkv_w, qkv_b, proj_w, proj_b, bias_t, relidx, comb, out);
  } else {
    hipLaunchKernelGGL((win_attn3d_kernel<false>), grid, block, 0, stream,
                       x, mask, qkv_w, qkv_b, proj_w, proj_b, bias_t, relidx,
                       (const float*)nullptr, out);
  }
}
